// EMN_18365280157750
// MI455X (gfx1250) — matrix operand order repaired and checked by reading, not compiled, not run on hardware
//
#include <hip/hip_runtime.h>
#include <hip/hip_bf16.h>
#include <math.h>

typedef __attribute__((ext_vector_type(16))) _Float16 v16h;
typedef __attribute__((ext_vector_type(8)))  float    v8f;

constexpr int kB    = 128;   // batches
constexpr int kN    = 64;    // nodes per graph
constexpr int kFN   = 64;    // node feature dim
constexpr int kFE   = 16;    // edge feature dim
constexpr int kH    = 128;   // hidden
constexpr int kOUT  = 32;    // output dim
constexpr int kKCAT = 144;   // FN+FN+FE
constexpr int kKPAD = 160;   // padded to 5 * 32 for WMMA K-tiles

// Branchless tanh: native v_tanh_f32 if the builtin exists on this toolchain,
// else 1 - 2*rcp(exp2(2*log2e*x)+1) (two trans ops, no EXEC-mask branches).
__device__ __forceinline__ float fast_tanh(float x) {
#if __has_builtin(__builtin_amdgcn_tanhf)
  return __builtin_amdgcn_tanhf(x);
#else
  float xc = fminf(fmaxf(x, -10.0f), 10.0f);
  float e  = __builtin_amdgcn_exp2f(xc * 2.8853900817779268f);  // 2*log2(e)
  return 1.0f - 2.0f * __builtin_amdgcn_rcpf(e + 1.0f);
#endif
}

// ---------------- WMMA fragment loaders (contiguous k = 16*(lane/16) + elem, same for A and B) ----

// A: 16x32 f16 tile at column k0 of an LDS row-major tile (ld = ldA halves).
// lanes 0-15: elems 0..15 = K0..15 ; lanes 16-31: K16..31
// Each 8-elem half is contiguous -> two 16B ds_load_b128.
__device__ __forceinline__ v16h load_a_frag_lds(const _Float16* At, int ldA, int k0, int lane) {
  int row = lane & 15;
  int kh  = (lane >> 4) << 4;
  const _Float16* p = At + row * ldA + k0 + kh;
  union { uint4 u[2]; v16h v; } r;
  r.u[0] = *(const uint4*)(p);        // K = k0+kh .. +7
  r.u[1] = *(const uint4*)(p + 8);    // K = k0+kh+8 .. +15
  return r.v;
}

// B: 32x16 f16 tile at (k0, n0) of an N-major (transposed) weight Wt[n][k], ld = ldK.
// lane col = lane%16, elems 0..15 = K (k0+kb)..(k0+kb+15): 32 contiguous bytes
// -> two global_load_b128, zero packing movs.
__device__ __forceinline__ v16h load_b_frag_t(const _Float16* __restrict__ Wt, int ldK,
                                              int k0, int n0, int lane) {
  int col = lane & 15;
  int kb  = (lane >> 4) << 4;
  const _Float16* p = Wt + (size_t)(n0 + col) * ldK + k0 + kb;
  union { uint4 u[2]; v16h v; } r;
  r.u[0] = *(const uint4*)(p);
  r.u[1] = *(const uint4*)(p + 8);
  return r.v;
}

// ---------------- K1: fp32 -> fp16 TRANSPOSED weight staging --------------------------
// W_emb (144x128) -> W_emb_t (128 x 160, K zero-padded); W_msg/W_edge (128x128) -> T.
__global__ void k_prep(const float* __restrict__ W_emb, const float* __restrict__ W_msg,
                       const float* __restrict__ W_edge,
                       _Float16* __restrict__ W_emb_t, _Float16* __restrict__ W_msg_t,
                       _Float16* __restrict__ W_edge_t) {
  const int total = kH * kKPAD + 2 * kH * kH;
  for (int i = blockIdx.x * blockDim.x + threadIdx.x; i < total; i += gridDim.x * blockDim.x) {
    if (i < kH * kKPAD) {
      int n = i / kKPAD, k = i % kKPAD;
      W_emb_t[i] = (k < kKCAT) ? (_Float16)W_emb[k * kH + n] : (_Float16)0.0f;
    } else if (i < kH * kKPAD + kH * kH) {
      int j = i - kH * kKPAD;
      int n = j / kH, k = j % kH;
      W_msg_t[j] = (_Float16)W_msg[k * kH + n];
    } else {
      int j = i - kH * kKPAD - kH * kH;
      int n = j / kH, k = j % kH;
      W_edge_t[j] = (_Float16)W_edge[k * kH + n];
    }
  }
}

// ---------------- K2: CSR offsets via binary search on sorted index arrays ------------
__device__ __forceinline__ int lbound(const int* __restrict__ arr, int len, int val) {
  int lo = 0, hi = len;
  while (lo < hi) { int mid = (lo + hi) >> 1; if (arr[mid] < val) lo = mid + 1; else hi = mid; }
  return lo;
}

__global__ void k_offsets(const int* __restrict__ igeb, int P,
                          const int* __restrict__ b_idx, int E,
                          int* __restrict__ e_off, int* __restrict__ b_off) {
  int t = blockIdx.x * blockDim.x + threadIdx.x;
  if (t <= E) e_off[t] = lbound(igeb, P, t);
  if (t <= kB) b_off[t] = lbound(b_idx, E, t);
}

// ---------------- K3: edge embedding  e_feat = tanh(cat @ W_emb + b_emb) --------------
// One block (8 waves) per 16-edge tile; each wave computes one 16x16 slice of H=128.
__global__ void k_embed(const float* __restrict__ nodes, const float* __restrict__ edges,
                        const int* __restrict__ b_idx, const int* __restrict__ n_idx,
                        const int* __restrict__ nhb_idx,
                        const _Float16* __restrict__ W_emb_t, const float* __restrict__ b_emb,
                        _Float16* __restrict__ e_feat_h, _Float16* __restrict__ memA, int E) {
  __shared__ _Float16 At[16 * kKPAD];
  const int m0 = blockIdx.x * 16;
  const int tid = threadIdx.x;

  // Gather cat = [nodes[b,n] | nodes[b,nhb] | edges[b,n,nhb] | 0-pad] into LDS (f16).
  {
    int row = tid >> 4;                 // 0..15
    int cbase = (tid & 15) * (kKPAD / 16);
    int e = m0 + row;
    int b = 0, n = 0, nh = 0;
    bool valid = (e < E);
    if (valid) { b = b_idx[e]; n = n_idx[e]; nh = nhb_idx[e]; }
#pragma unroll
    for (int j = 0; j < kKPAD / 16; ++j) {
      int c = cbase + j;
      float v = 0.0f;
      if (valid) {
        if (c < kFN)              v = nodes[((size_t)b * kN + n) * kFN + c];
        else if (c < 2 * kFN)     v = nodes[((size_t)b * kN + nh) * kFN + (c - kFN)];
        else if (c < kKCAT)       v = edges[(((size_t)b * kN + n) * kN + nh) * kFE + (c - 2 * kFN)];
      }
      At[row * kKPAD + c] = (_Float16)v;
    }
  }
  __syncthreads();

  const int wave = tid >> 5, lane = tid & 31;
  const int n0 = wave * 16;
  v8f acc = {};
#pragma unroll
  for (int kt = 0; kt < kKPAD / 32; ++kt) {
    v16h a  = load_a_frag_lds(At, kKPAD, kt * 32, lane);
    v16h bm = load_b_frag_t(W_emb_t, kKPAD, kt * 32, n0, lane);
    acc = __builtin_amdgcn_wmma_f32_16x16x32_f16(false, a, false, bm, (short)0, acc, false, false);
  }
  // Epilogue: bias + tanh, write e_feat (f16) and zero-init mem ping buffer.
  const int N  = lane & 15;
  const int mh = (lane >> 4) * 8;
  const float bias = b_emb[n0 + N];
#pragma unroll
  for (int r = 0; r < 8; ++r) {
    size_t idx = (size_t)(m0 + mh + r) * kH + n0 + N;
    e_feat_h[idx] = (_Float16)fast_tanh(acc[r] + bias);
    memA[idx]     = (_Float16)0.0f;
  }
}

// ---------------- K4: one message-passing round (gather fused with dual GEMM) ---------
// msg = segment_sum(mem_prev[eb]) ;  mem_next = tanh(msg@W_msg + e_feat@W_edge + b_prop)
__global__ void k_round(const _Float16* __restrict__ memPrev, const _Float16* __restrict__ e_feat_h,
                        const int* __restrict__ eb, const int* __restrict__ e_off,
                        const _Float16* __restrict__ W_msg_t, const _Float16* __restrict__ W_edge_t,
                        const float* __restrict__ b_prop, _Float16* __restrict__ memNext, int E) {
  __shared__ _Float16 Amsg[16 * kH];
  __shared__ _Float16 Aef [16 * kH];
  const int m0  = blockIdx.x * 16;
  const int tid = threadIdx.x;
  const int row = tid >> 4;     // 0..15
  const int cg  = tid & 15;     // 16B column group (8 halves)

  // Stage this tile's e_feat rows into LDS (16B vector copies).
  *((uint4*)(Aef + row * kH) + cg) =
      *((const uint4*)(e_feat_h + (size_t)(m0 + row) * kH) + cg);

  // Deterministic segmented gather-sum of incoming mem rows (f32 accumulate).
  {
    float acc8[8] = {0.f, 0.f, 0.f, 0.f, 0.f, 0.f, 0.f, 0.f};
    int e = m0 + row;
    if (e < E) {
      int k0 = e_off[e], k1 = e_off[e + 1];
      for (int k = k0; k < k1; ++k) {
        int src = eb[k];
        union { uint4 u; _Float16 h[8]; } pk;
        pk.u = *((const uint4*)(memPrev + (size_t)src * kH) + cg);
#pragma unroll
        for (int j = 0; j < 8; ++j) acc8[j] += (float)pk.h[j];
      }
    }
    union { uint4 u; _Float16 h[8]; } st;
#pragma unroll
    for (int j = 0; j < 8; ++j) st.h[j] = (_Float16)acc8[j];
    *((uint4*)(Amsg + row * kH) + cg) = st.u;
  }
  __syncthreads();

  const int wave = tid >> 5, lane = tid & 31;
  const int n0 = wave * 16;
  v8f acc = {};
#pragma unroll
  for (int kt = 0; kt < kH / 32; ++kt) {
    v16h a1 = load_a_frag_lds(Amsg, kH, kt * 32, lane);
    v16h b1 = load_b_frag_t(W_msg_t, kH, kt * 32, n0, lane);
    acc = __builtin_amdgcn_wmma_f32_16x16x32_f16(false, a1, false, b1, (short)0, acc, false, false);
    v16h a2 = load_a_frag_lds(Aef, kH, kt * 32, lane);
    v16h b2 = load_b_frag_t(W_edge_t, kH, kt * 32, n0, lane);
    acc = __builtin_amdgcn_wmma_f32_16x16x32_f16(false, a2, false, b2, (short)0, acc, false, false);
  }
  const int N  = lane & 15;
  const int mh = (lane >> 4) * 8;
  const float bias = b_prop[n0 + N];
#pragma unroll
  for (int r = 0; r < 8; ++r) {
    memNext[(size_t)(m0 + mh + r) * kH + n0 + N] = (_Float16)fast_tanh(acc[r] + bias);
  }
}

// ---------------- K5: per-batch pooling (node mask is implied by degree) + output GEMM
__global__ void k_final(const _Float16* __restrict__ memFin, const int* __restrict__ b_off,
                        const float* __restrict__ W_out, const float* __restrict__ b_out,
                        float* __restrict__ out) {
  __shared__ float g[kH];
  const int b = blockIdx.x;
  const int h = threadIdx.x;       // 128 threads
  float acc = 0.0f;
  const int e0 = b_off[b], e1 = b_off[b + 1];
  for (int e = e0; e < e1; ++e) acc += (float)memFin[(size_t)e * kH + h];
  g[h] = acc;
  __syncthreads();
  if (h < kOUT) {
    float r = b_out[h];
#pragma unroll 8
    for (int k = 0; k < kH; ++k) r += g[k] * W_out[k * kOUT + h];
    out[b * kOUT + h] = r;
  }
}

// ---------------- launch --------------------------------------------------------------
extern "C" void kernel_launch(void* const* d_in, const int* in_sizes, int n_in,
                              void* d_out, int out_size, void* d_ws, size_t ws_size,
                              hipStream_t stream) {
  const float* nodes  = (const float*)d_in[0];
  const float* edges  = (const float*)d_in[1];
  const float* W_emb  = (const float*)d_in[2];
  const float* b_emb  = (const float*)d_in[3];
  const float* W_msg  = (const float*)d_in[4];
  const float* W_edge = (const float*)d_in[5];
  const float* b_prop = (const float*)d_in[6];
  const float* W_out  = (const float*)d_in[7];
  const float* b_out  = (const float*)d_in[8];
  const int* e_bidx   = (const int*)d_in[9];
  const int* e_nidx   = (const int*)d_in[10];
  const int* e_nhbidx = (const int*)d_in[11];
  const int* ing_eb   = (const int*)d_in[12];
  const int* ing_igeb = (const int*)d_in[13];
  // d_in[14] (node_mask) is redundant: mask == (degree != 0), already implied by edges.

  const int E = in_sizes[9];
  const int P = in_sizes[12];
  const int Epad = (E + 15) & ~15;
  const int Etiles = Epad / 16;

  // Workspace carve-out (256B aligned slices).
  char* ws = (char*)d_ws;
  size_t off = 0;
  auto carve = [&](size_t bytes) -> void* {
    void* p = ws + off;
    off = (off + bytes + 255) & ~(size_t)255;
    return p;
  };
  _Float16* W_emb_t  = (_Float16*)carve((size_t)kH * kKPAD * sizeof(_Float16));
  _Float16* W_msg_t  = (_Float16*)carve((size_t)kH * kH * sizeof(_Float16));
  _Float16* W_edge_t = (_Float16*)carve((size_t)kH * kH * sizeof(_Float16));
  int*      e_off    = (int*)carve((size_t)(E + 1) * sizeof(int));
  int*      b_off    = (int*)carve((size_t)(kB + 1) * sizeof(int));
  _Float16* e_feat_h = (_Float16*)carve((size_t)Epad * kH * sizeof(_Float16));
  _Float16* memA     = (_Float16*)carve((size_t)Epad * kH * sizeof(_Float16));
  _Float16* memB     = (_Float16*)carve((size_t)Epad * kH * sizeof(_Float16));
  (void)n_in; (void)out_size; (void)ws_size;

  // 1) fp16 transposed weight staging
  {
    const int total = kH * kKPAD + 2 * kH * kH;
    k_prep<<<(total + 255) / 256, 256, 0, stream>>>(W_emb, W_msg, W_edge,
                                                    W_emb_t, W_msg_t, W_edge_t);
  }
  // 2) CSR offsets (incoming-edge segments + per-batch edge segments)
  k_offsets<<<(E + 1 + 255) / 256, 256, 0, stream>>>(ing_igeb, P, e_bidx, E, e_off, b_off);

  // 3) edge embedding (also zero-inits memA)
  k_embed<<<Etiles, 256, 0, stream>>>(nodes, edges, e_bidx, e_nidx, e_nhbidx,
                                      W_emb_t, b_emb, e_feat_h, memA, E);

  // 4) four message-passing rounds, ping-ponging mem buffers (final lands in memA)
  k_round<<<Etiles, 256, 0, stream>>>(memA, e_feat_h, ing_eb, e_off, W_msg_t, W_edge_t, b_prop, memB, E);
  k_round<<<Etiles, 256, 0, stream>>>(memB, e_feat_h, ing_eb, e_off, W_msg_t, W_edge_t, b_prop, memA, E);
  k_round<<<Etiles, 256, 0, stream>>>(memA, e_feat_h, ing_eb, e_off, W_msg_t, W_edge_t, b_prop, memB, E);
  k_round<<<Etiles, 256, 0, stream>>>(memB, e_feat_h, ing_eb, e_off, W_msg_t, W_edge_t, b_prop, memA, E);

  // 5) per-batch pooling + output projection
  k_final<<<kB, kH, 0, stream>>>(memA, b_off, W_out, b_out, (float*)d_out);
}
